// biggerbrain_15040975471264
// MI455X (gfx1250) — matrix operand order repaired and checked by reading, not compiled, not run on hardware
//
#include <hip/hip_runtime.h>

// ---------------------------------------------------------------------------
// Model constants (match reference)
// ---------------------------------------------------------------------------
#define VOCAB 50257
#define DIM   768
#define HEADS 8
#define DHEAD 96
#define SEQ   256
#define BATCH 8
#define FFNH  1536
#define NITER 3
#define TOK   (BATCH*SEQ)          // 2048

typedef __bf16 bf16_t;
typedef __attribute__((ext_vector_type(16))) __bf16 v16bf;
typedef __attribute__((ext_vector_type(8)))  __bf16 v8bf;
typedef __attribute__((ext_vector_type(4)))  __bf16 v4bf;
typedef __attribute__((ext_vector_type(8)))  float  v8f;

// ---------------------------------------------------------------------------
// Fragment load + dual WMMA for one K-step (shared by fast/edge loops).
// A-frag: lane&15 = row, lane>>4 selects K-half (ISA 7.12.2 16-bit A 16x32).
// B-frag: lane&15 = n, lane>>4 selects K-half (same K order as A); two adjacent 16-wide N subtiles.
// ---------------------------------------------------------------------------
__device__ __forceinline__ void gemm_step(const bf16_t* lA, const bf16_t* lB,
                                          int mt, int nth, int lane,
                                          v8f& acc0, v8f& acc1) {
  const int arow = mt * 16 + (lane & 15);
  const int kg   = (lane >> 4) * 8;        // 0 or 8
  const v8bf* ap = (const v8bf*)(lA + arow * 32);
  v8bf alo = ap[kg >> 3];                  // K = kg .. kg+7
  v8bf ahi = ap[(kg >> 3) + 2];            // K = kg+16 .. kg+23
  v16bf af;
#pragma unroll
  for (int i = 0; i < 8; ++i) { af[i] = alo[i]; af[8 + i] = ahi[i]; }
  const int bn = nth * 32 + (lane & 15);
  v16bf bf0, bf1;
#pragma unroll
  for (int i = 0; i < 8; ++i) {
    const int ka = kg + i, kb = kg + 16 + i;
    bf0[i] = lB[ka * 64 + bn]; bf0[8 + i] = lB[kb * 64 + bn];
    bf1[i] = lB[ka * 64 + bn + 16]; bf1[8 + i] = lB[kb * 64 + bn + 16];
  }
  acc0 = __builtin_amdgcn_wmma_f32_16x16x32_bf16(
      false, af, false, bf0, (short)0, acc0, false, false);
  acc1 = __builtin_amdgcn_wmma_f32_16x16x32_bf16(
      false, af, false, bf1, (short)0, acc1, false, false);
}

// ---------------------------------------------------------------------------
// Generic batched GEMM:  C[m,n] = sum_k A[m,k] * B'[k,n] (+bias[n]) (+=C)
//   btrans=1: B is row-major [N,K] (weights, y = x @ W^T)
//   btrans=0: B is row-major [K,N] (e.g. attn @ V)
// Tile: block = 256 threads = 8 waves; 64(M) x 64(N) per block, K-step 32.
// Each wave computes two 16x16 tiles via V_WMMA_F32_16X16X32_BF16.
// Interior blocks (all but the N-tail columns: vocab edge, 2nd block of N=96) take an unguarded
// float4-load / ds_store_b64 staging loop; edge blocks take a clamped-index
// loop (data selects, no EXEC divergence). Branch hoisted outside the K-loop.
// Batch: blockIdx.z = b*nh + h with independent strides for A/B/C.
// ---------------------------------------------------------------------------
__global__ __launch_bounds__(256) void k_gemm(
    const float* __restrict__ A, int lda, long sAb, long sAh,
    const float* __restrict__ Bm, int ldb, long sBb, long sBh, int btrans,
    const float* __restrict__ bias,
    float* __restrict__ C, int ldc, long sCb, long sCh,
    int M, int N, int K, int nh, int accumulate)
{
  __shared__ bf16_t lA[64 * 32];   // [m][k]
  __shared__ bf16_t lB[32 * 64];   // [k][n]

  const int tid  = threadIdx.x;
  const int zb   = blockIdx.z / nh;
  const int zh   = blockIdx.z % nh;
  const float* Ab = A  + (long)zb * sAb + (long)zh * sAh;
  const float* Bb = Bm + (long)zb * sBb + (long)zh * sBh;
  float*       Cb = C  + (long)zb * sCb + (long)zh * sCh;

  const int nbase = blockIdx.x * 64;
  const int mbase = blockIdx.y * 64;
  const int wave  = tid >> 5;
  const int lane  = tid & 31;
  const int mt    = wave >> 1;   // 0..3 (M subtile)
  const int nth   = wave & 1;    // 0..1 (pair of N subtiles)

  v8f acc0 = {}, acc1 = {};
  const int ksteps = K >> 5;     // all K used here are multiples of 32

  if ((mbase + 64 <= M) && (nbase + 64 <= N)) {
    // ================= interior fast path =================
    for (int kt = 0; kt < ksteps; ++kt) {
      const int k0 = kt << 5;
      // A tile: 64x32, two float4 per thread, coalesced over k
#pragma unroll
      for (int i = 0; i < 2; ++i) {
        int f = tid + i * 256;             // float4 index
        int r = f >> 3, cc = (f & 7) * 4;
        const float4 v = *(const float4*)(Ab + (long)(mbase + r) * lda + k0 + cc);
        v4bf p; p[0] = (bf16_t)v.x; p[1] = (bf16_t)v.y; p[2] = (bf16_t)v.z; p[3] = (bf16_t)v.w;
        *(v4bf*)(&lA[r * 32 + cc]) = p;
      }
      // B tile: 32(k) x 64(n)
      if (btrans) {                        // B[N,K]: float4 along k, scatter to [k][n]
#pragma unroll
        for (int i = 0; i < 2; ++i) {
          int f = tid + i * 256;
          int nn = f >> 3, kk = (f & 7) * 4;
          const float4 v = *(const float4*)(Bb + (long)(nbase + nn) * ldb + k0 + kk);
          lB[(kk + 0) * 64 + nn] = (bf16_t)v.x;
          lB[(kk + 1) * 64 + nn] = (bf16_t)v.y;
          lB[(kk + 2) * 64 + nn] = (bf16_t)v.z;
          lB[(kk + 3) * 64 + nn] = (bf16_t)v.w;
        }
      } else {                             // B[K,N]: float4 along n, contiguous in LDS
#pragma unroll
        for (int i = 0; i < 2; ++i) {
          int f = tid + i * 256;
          int kk = f >> 4, nn = (f & 15) * 4;
          const float4 v = *(const float4*)(Bb + (long)(k0 + kk) * ldb + nbase + nn);
          v4bf p; p[0] = (bf16_t)v.x; p[1] = (bf16_t)v.y; p[2] = (bf16_t)v.z; p[3] = (bf16_t)v.w;
          *(v4bf*)(&lB[kk * 64 + nn]) = p;
        }
      }
      __syncthreads();
      gemm_step(lA, lB, mt, nth, lane, acc0, acc1);
      __syncthreads();
    }
  } else {
    // ================= edge path (vocab / N=96 tail): clamped indices, no EXEC
    // divergence — loads always execute at a valid address, value selected.
    for (int kt = 0; kt < ksteps; ++kt) {
      const int k0 = kt << 5;
#pragma unroll
      for (int i = 0; i < 8; ++i) {
        int idx = tid + i * 256;
        int r = idx >> 5, cc = idx & 31;
        int gm  = mbase + r;
        int gmc = (gm < M) ? gm : (M - 1);
        float v = Ab[(long)gmc * lda + k0 + cc];
        lA[idx] = (bf16_t)((gm < M) ? v : 0.f);
      }
#pragma unroll
      for (int i = 0; i < 8; ++i) {
        int idx = tid + i * 256;
        int kk, nn; float v; int gn;
        if (btrans) {
          nn = idx >> 5; kk = idx & 31;
          gn = nbase + nn;
          int gnc = (gn < N) ? gn : (N - 1);
          v = Bb[(long)gnc * ldb + k0 + kk];
        } else {
          kk = idx >> 6; nn = idx & 63;
          gn = nbase + nn;
          int gnc = (gn < N) ? gn : (N - 1);
          v = Bb[(long)(k0 + kk) * ldb + gnc];
        }
        lB[kk * 64 + nn] = (bf16_t)((gn < N) ? v : 0.f);
      }
      __syncthreads();
      gemm_step(lA, lB, mt, nth, lane, acc0, acc1);
      __syncthreads();
    }
  }

  // ---- store: VGPR r = M row (lanes 16..31 carry M+8), lane&15 = N ----
  const int gn0   = nbase + nth * 32 + (lane & 15);
  const int gn1   = gn0 + 16;
  const int mrow0 = mbase + mt * 16 + ((lane >> 4) << 3);
  const float bv0 = (bias && gn0 < N) ? bias[gn0] : 0.f;
  const float bv1 = (bias && gn1 < N) ? bias[gn1] : 0.f;
#pragma unroll
  for (int r = 0; r < 8; ++r) {
    int gm = mrow0 + r;
    if (gm < M) {
      long rowoff = (long)gm * ldc;
      if (gn0 < N) {
        float val = acc0[r] + bv0;
        Cb[rowoff + gn0] = accumulate ? (Cb[rowoff + gn0] + val) : val;
      }
      if (gn1 < N) {
        float val = acc1[r] + bv1;
        Cb[rowoff + gn1] = accumulate ? (Cb[rowoff + gn1] + val) : val;
      }
    }
  }
}

// ---------------------------------------------------------------------------
// Elementwise / small kernels
// ---------------------------------------------------------------------------
__global__ void k_embed(const int* __restrict__ ids, const float* __restrict__ emb,
                        float* __restrict__ x) {
  int t = blockIdx.x;
  int c = blockIdx.y * 256 + threadIdx.x;
  x[(long)t * DIM + c] = emb[(long)ids[t] * DIM + c];
}

__global__ void k_fill(float* p, float v, long n) {
  long i = (long)blockIdx.x * 256 + threadIdx.x;
  if (i < n) p[i] = v;
}
__global__ void k_copy(float* __restrict__ d, const float* __restrict__ s, long n) {
  long i = (long)blockIdx.x * 256 + threadIdx.x;
  if (i < n) d[i] = s[i];
}
__global__ void k_add(float* __restrict__ d, const float* __restrict__ s, long n) {
  long i = (long)blockIdx.x * 256 + threadIdx.x;
  if (i < n) d[i] += s[i];
}

__global__ __launch_bounds__(256) void k_ln(const float* __restrict__ in,
                                            const float* __restrict__ w,
                                            const float* __restrict__ b,
                                            float* __restrict__ out) {
  __shared__ float red[256];
  const long base = (long)blockIdx.x * DIM;
  const int t = threadIdx.x;
  float v0 = in[base + t], v1 = in[base + t + 256], v2 = in[base + t + 512];
  red[t] = v0 + v1 + v2; __syncthreads();
  for (int s = 128; s > 0; s >>= 1) { if (t < s) red[t] += red[t + s]; __syncthreads(); }
  const float mean = red[0] * (1.f / DIM); __syncthreads();
  float d0 = v0 - mean, d1 = v1 - mean, d2 = v2 - mean;
  red[t] = d0 * d0 + d1 * d1 + d2 * d2; __syncthreads();
  for (int s = 128; s > 0; s >>= 1) { if (t < s) red[t] += red[t + s]; __syncthreads(); }
  const float rstd = rsqrtf(red[0] * (1.f / DIM) + 1e-5f);
  out[base + t]       = d0 * rstd * w[t]       + b[t];
  out[base + t + 256] = d1 * rstd * w[t + 256] + b[t + 256];
  out[base + t + 512] = d2 * rstd * w[t + 512] + b[t + 512];
}

// RoPE: one thread handles the (i2, i2+48) pair so no cross-block race.
__global__ void k_rope(float* __restrict__ q, float* __restrict__ k) {
  int t = blockIdx.x;                            // token
  int j = blockIdx.y * 128 + threadIdx.x;        // 0..383
  int s  = t & (SEQ - 1);
  int h  = j / 48, i2 = j % 48;
  float freq = __powf(10000.f, -(float)(2 * i2) / (float)DHEAD);
  float ang = (float)s * freq;
  float cs = __cosf(ang), sn = __sinf(ang);
  long base = (long)t * DIM + h * DHEAD;
  float q0 = q[base + i2], q1 = q[base + i2 + 48];
  q[base + i2]      = q0 * cs - q1 * sn;
  q[base + i2 + 48] = q1 * cs + q0 * sn;
  float k0 = k[base + i2], k1 = k[base + i2 + 48];
  k[base + i2]      = k0 * cs - k1 * sn;
  k[base + i2 + 48] = k1 * cs + k0 * sn;
}

__global__ __launch_bounds__(256) void k_softmax(float* __restrict__ sc, float scale) {
  __shared__ float red[256];
  const int bh = blockIdx.y, row = blockIdx.x, t = threadIdx.x;
  const long base = ((long)bh * SEQ + row) * SEQ;
  float v = (t <= row) ? sc[base + t] * scale : -3.0e38f;
  red[t] = v; __syncthreads();
  for (int s = 128; s > 0; s >>= 1) { if (t < s) red[t] = fmaxf(red[t], red[t + s]); __syncthreads(); }
  const float mx = red[0]; __syncthreads();
  float e = (t <= row) ? __expf(v - mx) : 0.f;
  red[t] = e; __syncthreads();
  for (int s = 128; s > 0; s >>= 1) { if (t < s) red[t] += red[t + s]; __syncthreads(); }
  sc[base + t] = e / red[0];
}

__global__ void k_silu_mul(const float* __restrict__ hb, float* __restrict__ g, int hid) {
  int row = blockIdx.x;
  int c = blockIdx.y * 256 + threadIdx.x;
  float a  = hb[(long)row * 2 * hid + c];
  float b2 = hb[(long)row * 2 * hid + hid + c];
  g[(long)row * hid + c] = a / (1.f + __expf(-a)) * b2;
}

// GRU scan: one block per batch; h-state in LDS. Thread t owns rows t+256m of
// w_hh so hr/hz/hn for its 3 output columns are all local (no exchange).
__global__ __launch_bounds__(256) void k_gru(const float* __restrict__ gi,
                                             const float* __restrict__ whh,
                                             const float* __restrict__ bhh,
                                             float* __restrict__ outs,
                                             float* __restrict__ mem) {
  __shared__ float h[DIM];
  const int b = blockIdx.x, t = threadIdx.x;
  h[t] = mem[b * DIM + t]; h[t + 256] = mem[b * DIM + t + 256]; h[t + 512] = mem[b * DIM + t + 512];
  __syncthreads();
  for (int s = 0; s < SEQ; ++s) {
    float gh[9];
#pragma unroll
    for (int m = 0; m < 9; ++m) {
      int n = t + m * 256;
      const float* wr = whh + (long)n * DIM;
      float acc = bhh[n];
      for (int kk = 0; kk < DIM; ++kk) acc += h[kk] * wr[kk];
      gh[m] = acc;
    }
    __syncthreads();
    const long gbase = (long)(b * SEQ + s) * (3 * DIM);
    const long obase = (long)(b * SEQ + s) * DIM;
#pragma unroll
    for (int m = 0; m < 3; ++m) {
      int j = t + m * 256;
      float ir  = gi[gbase + j];
      float iz  = gi[gbase + DIM + j];
      float inn = gi[gbase + 2 * DIM + j];
      float r  = 1.f / (1.f + __expf(-(ir + gh[m])));
      float zz = 1.f / (1.f + __expf(-(iz + gh[m + 3])));
      float nn = tanhf(inn + r * gh[m + 6]);
      float hn = (1.f - zz) * nn + zz * h[j];
      outs[obase + j] = hn;
      h[j] = hn;
    }
    __syncthreads();
  }
  mem[b * DIM + t] = h[t]; mem[b * DIM + t + 256] = h[t + 256]; mem[b * DIM + t + 512] = h[t + 512];
}

// gated residual: g = sigmoid([x,res,x-res] . wg); out = x*g + res*(1-g)
__global__ __launch_bounds__(256) void k_gated(const float* __restrict__ x,
                                               const float* __restrict__ res,
                                               const float* __restrict__ wg,
                                               float* __restrict__ out) {
  __shared__ float red[256];
  const int tok = blockIdx.x, t = threadIdx.x;
  const long base = (long)tok * DIM;
  float acc = 0.f;
#pragma unroll
  for (int m = 0; m < 9; ++m) {
    int j = t + m * 256;
    float v;
    if (j < DIM)          v = x[base + j];
    else if (j < 2 * DIM) v = res[base + j - DIM];
    else                  v = x[base + j - 2 * DIM] - res[base + j - 2 * DIM];
    acc += v * wg[j];
  }
  red[t] = acc; __syncthreads();
  for (int s = 128; s > 0; s >>= 1) { if (t < s) red[t] += red[t + s]; __syncthreads(); }
  const float g = 1.f / (1.f + __expf(-red[0]));
#pragma unroll
  for (int m = 0; m < 3; ++m) {
    int j = t + m * 256;
    out[base + j] = x[base + j] * g + res[base + j] * (1.f - g);
  }
}

__global__ void k_total_add(float* total, const float* active) {
  int t = threadIdx.x; if (t < BATCH) total[t] += active[t];
}
__global__ void k_select(float* __restrict__ y, const float* __restrict__ yp,
                         const float* __restrict__ active) {
  int tok = blockIdx.x;
  int c = blockIdx.y * 256 + threadIdx.x;
  if (active[tok >> 8] <= 0.f) y[(long)tok * DIM + c] = yp[(long)tok * DIM + c];
}
__global__ __launch_bounds__(256) void k_exit(const float* __restrict__ y,
                                              const float* __restrict__ pw,
                                              float* __restrict__ active) {
  __shared__ float red[256];
  const int b = blockIdx.x, t = threadIdx.x;
  const long base = (long)(b * SEQ + SEQ - 1) * DIM;
  red[t] = y[base + t] * pw[t] + y[base + t + 256] * pw[t + 256] + y[base + t + 512] * pw[t + 512];
  __syncthreads();
  for (int s = 128; s > 0; s >>= 1) { if (t < s) red[t] += red[t + s]; __syncthreads(); }
  if (t == 0) {
    float p = 1.f / (1.f + __expf(-red[0]));
    active[b] *= (p < 0.5f) ? 1.f : 0.f;
  }
}
__global__ void k_store_total(float* out, const float* total) {
  int t = threadIdx.x; if (t < BATCH) out[(long)TOK * VOCAB + t] = total[t];
}

// ---------------------------------------------------------------------------
// Host-side orchestration
// ---------------------------------------------------------------------------
struct Ctx {
  hipStream_t st;
  float *x, *y, *z, *yn, *ao, *q, *k, *v, *cx, *m1, *yp, *gi, *sc, *h, *g, *mem, *act, *tot;
};
struct AttnW { const float *wq,*bq,*wk,*bk,*wv,*bv,*wo,*bo; };

static inline void gemm(const Ctx& c,
    const float* A, int lda, long sAb, long sAh,
    const float* B, int ldb, long sBb, long sBh, int btrans,
    const float* bias, float* C, int ldc, long sCb, long sCh,
    int M, int N, int K, int nb, int nh, int accum) {
  dim3 grid((N + 63) / 64, (M + 63) / 64, nb * nh);
  k_gemm<<<grid, 256, 0, c.st>>>(A, lda, sAb, sAh, B, ldb, sBb, sBh, btrans,
                                 bias, C, ldc, sCb, sCh, M, N, K, nh, accum);
}
static inline void linear(const Ctx& c, const float* x, const float* W,
                          const float* b, float* out, int N, int K) {
  gemm(c, x, K, 0, 0, W, K, 0, 0, 1, b, out, N, 0, 0, TOK, N, K, 1, 1, 0);
}
static inline void eadd(const Ctx& c, float* d, const float* s) {
  k_add<<<(TOK * DIM) / 256, 256, 0, c.st>>>(d, s, (long)TOK * DIM);
}
static inline void ecopy(const Ctx& c, float* d, const float* s) {
  k_copy<<<(TOK * DIM) / 256, 256, 0, c.st>>>(d, s, (long)TOK * DIM);
}
static inline void lnorm(const Ctx& c, const float* in, const float* w,
                         const float* b, float* out) {
  k_ln<<<TOK, 256, 0, c.st>>>(in, w, b, out);
}

// full multi-head attention core given q,k,v buffers filled ([TOK,DIM], heads
// in contiguous 96-col slices); writes [TOK,DIM] context @ Wo + bo -> out
static void attn_core(const Ctx& c, const float* wo, const float* bo, float* out) {
  // scores[b,h] = q @ k^T    (per-head 256x256x96)
  gemm(c, c.q, DIM, (long)SEQ * DIM, DHEAD,
          c.k, DIM, (long)SEQ * DIM, DHEAD, 1, nullptr,
          c.sc, SEQ, (long)HEADS * SEQ * SEQ, (long)SEQ * SEQ,
          SEQ, SEQ, DHEAD, BATCH, HEADS, 0);
  k_softmax<<<dim3(SEQ, BATCH * HEADS), 256, 0, c.st>>>(c.sc, 0.1020620726f); // 1/sqrt(96)
  // ctx[b,h] = scores @ v    (per-head 256x96x256)
  gemm(c, c.sc, SEQ, (long)HEADS * SEQ * SEQ, (long)SEQ * SEQ,
          c.v, DIM, (long)SEQ * DIM, DHEAD, 0, nullptr,
          c.cx, DIM, (long)SEQ * DIM, DHEAD,
          SEQ, DHEAD, SEQ, BATCH, HEADS, 0);
  linear(c, c.cx, wo, bo, out, DIM, DIM);
}

static void rope_attn(const Ctx& c, const AttnW& w, const float* xin, float* out) {
  linear(c, xin, w.wq, w.bq, c.q, DIM, DIM);
  linear(c, xin, w.wk, w.bk, c.k, DIM, DIM);
  linear(c, xin, w.wv, w.bv, c.v, DIM, DIM);
  k_rope<<<dim3(TOK, 3), 128, 0, c.st>>>(c.q, c.k);
  attn_core(c, w.wo, w.bo, out);
}

static void mha(const Ctx& c, const float* in_w, const float* in_b,
                const float* out_w, const float* out_b,
                const float* qin, const float* kv, float* out) {
  linear(c, qin, in_w,                  in_b,           c.q, DIM, DIM);
  linear(c, kv,  in_w + (long)DIM*DIM,  in_b + DIM,     c.k, DIM, DIM);
  linear(c, kv,  in_w + 2L*DIM*DIM,     in_b + 2*DIM,   c.v, DIM, DIM);
  attn_core(c, out_w, out_b, out);
}

static void swiglu_ffn(const Ctx& c, const float* xin, const float* w1,
                       const float* b1, const float* w2, const float* b2,
                       int hid, float* add_target) {
  linear(c, xin, w1, b1, c.h, 2 * hid, DIM);
  k_silu_mul<<<dim3(TOK, hid / 256), 256, 0, c.st>>>(c.h, c.g, hid);
  linear(c, c.g, w2, b2, c.ao, DIM, hid);
  eadd(c, add_target, c.ao);
}

static void gru(const Ctx& c, const float* w_ih, const float* b_ih,
                const float* w_hh, const float* b_hh, const float* xin) {
  linear(c, xin, w_ih, b_ih, c.gi, 3 * DIM, DIM);
  k_gru<<<BATCH, 256, 0, c.st>>>(c.gi, w_hh, b_hh, c.m1, c.mem);
}

extern "C" void kernel_launch(void* const* d_in, const int* in_sizes, int n_in,
                              void* d_out, int out_size, void* d_ws, size_t ws_size,
                              hipStream_t stream) {
  // -------- input flattening: jax pytree order (dict keys sorted, recursive) --
  // 0: input_ids
  // attn blocks A1,A2,A3,A4,PostA,PreA; each: bk,bo,bq,bv,wk,wo,wq,wv (1..48)
  // 49: embed
  // ffn_post: 50 b1, 51 b2, 52 w1, 53 w2 ; ffn_pre: 54..57
  // 58: gate_w
  // gru: 59 b_hh, 60 b_ih, 61 w_hh, 62 w_ih
  // ln (b,w pairs): n0 63/64 n1 65/66 n2 67/68 n3 69/70 n4 71/72 post 73/74 pre 75/76
  // mha: 77 in_b, 78 in_w, 79 out_b, 80 out_w
  // mi Mi1..Mi4 each b1,b2,w1,w2: 81..96
  // 97: ponder_w
  const int* ids = (const int*)d_in[0];
  auto F = [&](int i) { return (const float*)d_in[i]; };

  auto attnw = [&](int blk) {  // blk: 0=A1 1=A2 2=A3 3=A4 4=PostA 5=PreA
    int b0 = 1 + blk * 8;
    AttnW w;
    w.bk = F(b0 + 0); w.bo = F(b0 + 1); w.bq = F(b0 + 2); w.bv = F(b0 + 3);
    w.wk = F(b0 + 4); w.wo = F(b0 + 5); w.wq = F(b0 + 6); w.wv = F(b0 + 7);
    return w;
  };
  AttnW A1 = attnw(0), A2 = attnw(1), A3 = attnw(2), A4 = attnw(3);
  AttnW PostA = attnw(4), PreA = attnw(5);
  const float* embed = F(49);
  const float *fpost_b1 = F(50), *fpost_b2 = F(51), *fpost_w1 = F(52), *fpost_w2 = F(53);
  const float *fpre_b1  = F(54), *fpre_b2  = F(55), *fpre_w1  = F(56), *fpre_w2  = F(57);
  const float* gate_w = F(58);
  const float *gru_bhh = F(59), *gru_bih = F(60), *gru_whh = F(61), *gru_wih = F(62);
  const float* ln_b[7]; const float* ln_w[7];   // n0..n4, post, pre
  for (int i = 0; i < 7; ++i) { ln_b[i] = F(63 + 2 * i); ln_w[i] = F(64 + 2 * i); }
  const float *mha_inb = F(77), *mha_inw = F(78), *mha_outb = F(79), *mha_outw = F(80);
  const float *mi_b1[4], *mi_b2[4], *mi_w1[4], *mi_w2[4];
  for (int i = 0; i < 4; ++i) {
    mi_b1[i] = F(81 + 4 * i); mi_b2[i] = F(82 + 4 * i);
    mi_w1[i] = F(83 + 4 * i); mi_w2[i] = F(84 + 4 * i);
  }
  const float* ponder_w = F(97);

  // -------- workspace layout (floats) --------
  const long TD = (long)TOK * DIM;
  float* w = (float*)d_ws;
  Ctx c; c.st = stream;
  long off = 0;
  c.x  = w + off; off += TD;
  c.y  = w + off; off += TD;
  c.z  = w + off; off += TD;
  c.yn = w + off; off += TD;
  c.ao = w + off; off += TD;
  c.q  = w + off; off += TD;
  c.k  = w + off; off += TD;
  c.v  = w + off; off += TD;
  c.cx = w + off; off += TD;
  c.m1 = w + off; off += TD;
  c.yp = w + off; off += TD;
  c.gi = w + off; off += (long)TOK * 3 * DIM;
  c.sc = w + off; off += (long)BATCH * HEADS * SEQ * SEQ;
  c.h  = w + off; off += (long)TOK * 2 * 4 * DIM;    // up to 2*hid = 6144
  c.g  = w + off; off += (long)TOK * 4 * DIM;        // up to hid = 3072
  c.mem = w + off; off += (long)BATCH * DIM;
  c.act = w + off; off += 32;
  c.tot = w + off; off += 32;
  if (ws_size < (size_t)off * sizeof(float)) return;  // cannot allocate; bail

  // -------- init state --------
  k_fill<<<(BATCH * DIM + 255) / 256, 256, 0, stream>>>(c.mem, 0.f, BATCH * DIM);
  k_fill<<<1, 32, 0, stream>>>(c.act, 1.f, BATCH);
  k_fill<<<1, 32, 0, stream>>>(c.tot, 0.f, BATCH);

  // -------- forward --------
  k_embed<<<dim3(TOK, 3), 256, 0, stream>>>(ids, embed, c.x);

  rope_attn(c, A1, c.x, c.y);                               // y = attn(A1, x)
  gru(c, gru_wih, gru_bih, gru_whh, gru_bhh, c.y);          // m1, mem
  ecopy(c, c.z, c.y);                                       // z = y
  lnorm(c, c.y, ln_w[6], ln_b[6], c.y);                     // y = ln(y, pre)
  rope_attn(c, PreA, c.y, c.ao); eadd(c, c.y, c.ao);        // y = attn(PreA,y)+y
  swiglu_ffn(c, c.y, fpre_w1, fpre_b1, fpre_w2, fpre_b2, FFNH, c.y);

  for (int it = 0; it < NITER; ++it) {
    ecopy(c, c.yp, c.y);                                    // y_prev
    lnorm(c, c.y, ln_w[0], ln_b[0], c.yn);                  // yn = ln(y, n0)
    rope_attn(c, A1, c.yn, c.ao); eadd(c, c.y, c.ao);       // y = attn(A1,yn)+y
    mha(c, mha_inw, mha_inb, mha_outw, mha_outb, c.y, c.m1, c.y); // y = mha(y,m1)
    gru(c, gru_wih, gru_bih, gru_whh, gru_bhh, c.y);        // m1, mem
    lnorm(c, c.y, ln_w[1], ln_b[1], c.y);                   // y = ln(y, n1)
    swiglu_ffn(c, c.y, mi_w1[0], mi_b1[0], mi_w2[0], mi_b2[0], 3072, c.y);
    lnorm(c, c.y, ln_w[2], ln_b[2], c.y);                   // n2
    rope_attn(c, A2, c.y, c.ao); eadd(c, c.y, c.ao);
    swiglu_ffn(c, c.y, mi_w1[1], mi_b1[1], mi_w2[1], mi_b2[1], 3072, c.y);
    lnorm(c, c.y, ln_w[3], ln_b[3], c.y);                   // n3
    rope_attn(c, A3, c.y, c.ao); eadd(c, c.y, c.ao);
    swiglu_ffn(c, c.y, mi_w1[2], mi_b1[2], mi_w2[2], mi_b2[2], 3072, c.y);
    lnorm(c, c.y, ln_w[4], ln_b[4], c.y);                   // n4
    rope_attn(c, A4, c.y, c.ao); eadd(c, c.y, c.ao);
    swiglu_ffn(c, c.y, mi_w1[3], mi_b1[3], mi_w2[3], mi_b2[3], 3072, c.y);
    k_gated<<<TOK, 256, 0, stream>>>(c.y, c.z, gate_w, c.y);
    k_total_add<<<1, 32, 0, stream>>>(c.tot, c.act);
    k_select<<<dim3(TOK, 3), 256, 0, stream>>>(c.y, c.yp, c.act);
    k_exit<<<BATCH, 256, 0, stream>>>(c.y, ponder_w, c.act);
    ecopy(c, c.z, c.y);
  }

  lnorm(c, c.z, ln_w[5], ln_b[5], c.z);                     // z = ln(z, post)
  rope_attn(c, PostA, c.z, c.ao); eadd(c, c.z, c.ao);
  swiglu_ffn(c, c.z, fpost_w1, fpost_b1, fpost_w2, fpost_b2, FFNH, c.z);

  // logits = z @ embed^T   [TOK, VOCAB] into d_out
  float* out = (float*)d_out;
  gemm(c, c.z, DIM, 0, 0, embed, DIM, 0, 0, 1, nullptr,
       out, VOCAB, 0, 0, TOK, VOCAB, DIM, 1, 1, 0);
  k_store_total<<<1, 32, 0, stream>>>(out, c.tot);
  (void)in_sizes; (void)n_in; (void)out_size;
}
